// Face_Rotate_Conv_80985903334293
// MI455X (gfx1250) — matrix operand order repaired and checked by reading, not compiled, not run on hardware
//
#include <hip/hip_runtime.h>
#include <hip/hip_bf16.h>

typedef __attribute__((ext_vector_type(16))) _Float16 v16h;
typedef __attribute__((ext_vector_type(8)))  float    v8f;

union H16 { v16h v; _Float16 h[16]; unsigned int u[8]; };
union H8  { unsigned int u[4]; _Float16 h[8]; };

struct F3 { float x, y, z; };   // 4-byte aligned 12B chunk -> global_load_b96

// Pack two f32 -> one dword of packed f16 (v_cvt_pk_rtz, single VALU op, no canonicalize)
__device__ __forceinline__ unsigned int pk2(float a, float b) {
    return __builtin_bit_cast(unsigned int, __builtin_amdgcn_cvt_pkrtz(a, b));
}
// Packed-f16 ReLU: one VOP3P instruction clamps both halves at 0.
__device__ __forceinline__ unsigned int pkmax0(unsigned int x) {
    unsigned int r;
    asm("v_pk_max_num_f16 %0, %1, 0" : "=v"(r) : "v"(x));
    return r;
}

// f16 A-matrix (16xK=32) per-lane K index for half i (0..15), half-wave group g:
// lanes 0-15:  halves 0-15 -> K=0..15
// lanes 16-31: halves 0-15 -> K=16..31   (contiguous, matches B2 packing)
__device__ __forceinline__ int akidx(int i, int g) {
    return i + 16 * g;
}

// ---------------- prep: fold conv(+mean)+w3 into a 3x64 affine map, f32->f16 ----
// ws layout:  [0, 4096)    m1t  : _Float16[64][32]  (M1^T rows j, cols k; k>=3 zero)
//             [4096,12288) w4h  : _Float16[64][64]  (w4 row-major, f16)
//             [12288,12544) bias1 : float[64]
__global__ void prep_fold(const float* __restrict__ conv_w, const float* __restrict__ conv_b,
                          const float* __restrict__ w3, const float* __restrict__ b3,
                          const float* __restrict__ w4,
                          _Float16* __restrict__ m1t, _Float16* __restrict__ w4h,
                          float* __restrict__ bias1) {
    int j = threadIdx.x;
    if (j >= 64) return;
    float b = b3[j];
    float m0 = 0.f, m1 = 0.f, m2 = 0.f;
    const float inv3 = 1.0f / 3.0f;
    for (int o = 0; o < 32; ++o) {
        float w = w3[j * 32 + o];          // w3 is (64,32) row-major
        b += conv_b[o] * w;
        m0 += w * (conv_w[o * 6 + 0] + conv_w[o * 6 + 3]) * inv3;
        m1 += w * (conv_w[o * 6 + 1] + conv_w[o * 6 + 4]) * inv3;
        m2 += w * (conv_w[o * 6 + 2] + conv_w[o * 6 + 5]) * inv3;
    }
    bias1[j] = b;
    for (int k = 0; k < 32; ++k) {
        float mv = (k == 0) ? m0 : (k == 1) ? m1 : (k == 2) ? m2 : 0.f;
        m1t[j * 32 + k] = (_Float16)mv;
    }
    for (int k = 0; k < 64; ++k) w4h[j * 64 + k] = (_Float16)w4[j * 64 + k];
}

// ---------------- main: 16 faces per wave-iteration, all-WMMA, no LDS ----------
__global__ __launch_bounds__(256) void face_mlp_wmma(
    const float* __restrict__ corner,
    const _Float16* __restrict__ m1t, const float* __restrict__ bias1,
    const _Float16* __restrict__ w4h, const float* __restrict__ b4,
    float* __restrict__ out, int nFaces, int nTiles, int nWaves) {

    const int lane = threadIdx.x & 31;
    const int g    = lane >> 4;       // half-wave group (C/D: M = v + 8g)
    const int m    = lane & 15;       // face-in-tile == D's N index
    const int wave = (int)((blockIdx.x * blockDim.x + threadIdx.x) >> 5);

    // ---- loop-invariant A-operands (weights) and bias accumulator images ----
    H16 a1[4];                        // M1^T tiles: 16 j-rows x K=32
    #pragma unroll
    for (int jt = 0; jt < 4; ++jt) {
        int j = 16 * jt + m;
        #pragma unroll
        for (int i = 0; i < 16; ++i) a1[jt].h[i] = m1t[j * 32 + akidx(i, g)];
    }
    H16 a2[4][2];                     // w4 tiles: 16 jout-rows x K=64 (2 chunks)
    #pragma unroll
    for (int jt = 0; jt < 4; ++jt)
        #pragma unroll
        for (int c = 0; c < 2; ++c)
            #pragma unroll
            for (int i = 0; i < 16; ++i)
                a2[jt][c].h[i] = w4h[(16 * jt + m) * 64 + 32 * c + akidx(i, g)];

    v8f b1c[4], b4c[4];               // biases pre-spread into C-layout (M = v+8g)
    #pragma unroll
    for (int jt = 0; jt < 4; ++jt) {
        const float* p1 = bias1 + 16 * jt + 8 * g;
        const float* p4 = b4    + 16 * jt + 8 * g;
        #pragma unroll
        for (int v = 0; v < 8; ++v) { b1c[jt][v] = p1[v]; b4c[jt][v] = p4[v]; }
    }

    // B1 register image hoisted: K rows 3..31 are permanently zero.
    H16 b1;
    #pragma unroll
    for (int d = 0; d < 8; ++d) b1.u[d] = 0u;

    for (int tile = wave; tile < nTiles; tile += nWaves) {
        const int face = tile * 16 + m;

        // t = c1 + c2 + c3 ; only group-0 lanes feed B1 (K=16..31 rows are zero).
        // g==1 lanes keep t==0, so the packed b1 dwords need no per-lane selects.
        float t0 = 0.f, t1 = 0.f, t2 = 0.f;
        if (g == 0) {
            const F3* cp = (const F3*)(corner + (size_t)min(face, nFaces - 1) * 9);
            F3 c1 = cp[0], c2 = cp[1], c3 = cp[2];
            t0 = c1.x + c2.x + c3.x;
            t1 = c1.y + c2.y + c3.y;
            t2 = c1.z + c2.z + c3.z;
        }
        // Branch-free prefetch of next tile's corner rows (global_prefetch_b8):
        // clamp keeps the address in-bounds so no EXEC-mask juggling is needed.
        {
            int pfFace = min(tile + nWaves, nTiles - 1) * 16 + m;
            __builtin_prefetch(corner + (size_t)pfFace * 9, 0, 0);
        }
        b1.u[0] = pk2(t0, t1);
        b1.u[1] = pk2(t2, 0.f);

        // Layer A: h.T tiles = M1^T x t.T + bias1 (4 WMMAs)
        v8f d1[4];
        #pragma unroll
        for (int jt = 0; jt < 4; ++jt)
            d1[jt] = __builtin_amdgcn_wmma_f32_16x16x32_f16(
                false, a1[jt].v, false, b1.v, (short)0, b1c[jt], false, false);

        // Pack to f16 (cvt_pk_rtz) then ReLU both halves in one VOP3P (v_pk_max 0)
        H8 own[4];
        #pragma unroll
        for (int jt = 0; jt < 4; ++jt)
            #pragma unroll
            for (int d = 0; d < 4; ++d)
                own[jt].u[d] = pkmax0(pk2(d1[jt][2 * d], d1[jt][2 * d + 1]));

        // Half-wave exchange (partner lane = lane^16, same face): pre-select the
        // single tile the partner needs, permute it, select on receive -> 8 bpermutes.
        //   partner of g=0 wants our own[2c+1]; partner of g=1 wants our own[2c]
        H8 rcv[2];
        #pragma unroll
        for (int c = 0; c < 2; ++c)
            #pragma unroll
            for (int d = 0; d < 4; ++d) {
                unsigned int src = g ? own[2 * c].u[d] : own[2 * c + 1].u[d];
                rcv[c].u[d] = __shfl_xor(src, 16, 32);
            }

        // B2 k-chunks (32x16 each):
        //   g=0: K=0..7 own[2c], K=8..15 rcv[c](=partner own[2c])
        //   g=1: K=16..23 rcv[c](=partner own[2c+1]), K=24..31 own[2c+1]
        H16 b2[2];
        #pragma unroll
        for (int c = 0; c < 2; ++c)
            #pragma unroll
            for (int d = 0; d < 4; ++d) {
                b2[c].u[d]     = g ? rcv[c].u[d]        : own[2 * c].u[d];
                b2[c].u[4 + d] = g ? own[2 * c + 1].u[d] : rcv[c].u[d];
            }

        // Layer B: out.T tiles = w4 x h.T + b4 (8 WMMAs), streaming (NT) stores
        float* op = out + (size_t)face * 64 + 8 * g;
        const bool st = (face < nFaces);
        #pragma unroll
        for (int jt = 0; jt < 4; ++jt) {
            v8f acc = b4c[jt];
            acc = __builtin_amdgcn_wmma_f32_16x16x32_f16(
                false, a2[jt][0].v, false, b2[0].v, (short)0, acc, false, false);
            acc = __builtin_amdgcn_wmma_f32_16x16x32_f16(
                false, a2[jt][1].v, false, b2[1].v, (short)0, acc, false, false);
            if (st) __builtin_nontemporal_store(acc, (v8f*)(op + 16 * jt));
        }
    }
}

extern "C" void kernel_launch(void* const* d_in, const int* in_sizes, int n_in,
                              void* d_out, int out_size, void* d_ws, size_t ws_size,
                              hipStream_t stream) {
    const float* corner = (const float*)d_in[0];
    const float* conv_w = (const float*)d_in[1];
    const float* conv_b = (const float*)d_in[2];
    const float* w3     = (const float*)d_in[3];
    const float* b3     = (const float*)d_in[4];
    const float* w4     = (const float*)d_in[5];
    const float* b4     = (const float*)d_in[6];
    float* out = (float*)d_out;

    const int nFaces = in_sizes[0] / 9;
    const int nTiles = (nFaces + 15) / 16;

    _Float16* m1t   = (_Float16*)d_ws;                     // 4 KB
    _Float16* w4h   = (_Float16*)((char*)d_ws + 4096);     // 8 KB
    float*    bias1 = (float*)((char*)d_ws + 12288);       // 256 B

    prep_fold<<<1, 64, 0, stream>>>(conv_w, conv_b, w3, b3, w4, m1t, w4h, bias1);

    const int threads = 256;                 // 8 waves / block
    int blocks = (nTiles + 7) / 8;
    if (blocks > 2048) blocks = 2048;
    if (blocks < 1) blocks = 1;
    const int nWaves = blocks * (threads / 32);

    face_mlp_wmma<<<blocks, threads, 0, stream>>>(
        corner, m1t, bias1, w4h, b4, out, nFaces, nTiles, nWaves);
}
